// KimiDeltaAttention_68195490726078
// MI455X (gfx1250) — compile-verified, matrix operand order checked by reading, not run on hardware
//
#include <hip/hip_runtime.h>
#include <hip/hip_bf16.h>
#include <math.h>

// Problem constants (match reference)
#define BSZ   2
#define TLEN  2048
#define HIDD  2048
#define NH    16
#define HD    128
#define KC    4
#define EPSV  1e-6f

typedef __bf16 bf16_t;
typedef __attribute__((ext_vector_type(16))) __bf16 v16bf;
typedef __attribute__((ext_vector_type(8)))  __bf16 v8bf;
typedef __attribute__((ext_vector_type(8)))  float  v8f;

// ---- CDNA5 async global->LDS copies (ASYNCcnt) ----------------------------
__device__ __forceinline__ void async_lds_b128(unsigned lds_addr, const void* g) {
  asm volatile("global_load_async_to_lds_b128 %0, %1, off"
               :: "v"(lds_addr), "v"((unsigned long long)(size_t)g)
               : "memory");
}
__device__ __forceinline__ void async_lds_b64(unsigned lds_addr, const void* g) {
  asm volatile("global_load_async_to_lds_b64 %0, %1, off"
               :: "v"(lds_addr), "v"((unsigned long long)(size_t)g)
               : "memory");
}
__device__ __forceinline__ void async_lds_b32(unsigned lds_addr, const void* g) {
  asm volatile("global_load_async_to_lds_b32 %0, %1, off"
               :: "v"(lds_addr), "v"((unsigned long long)(size_t)g)
               : "memory");
}
__device__ __forceinline__ void wait_async_le0() {
  asm volatile("s_wait_asynccnt 0x0" ::: "memory");
}
__device__ __forceinline__ void wait_async_le4() {
  asm volatile("s_wait_asynccnt 0x4" ::: "memory");
}

// ---------------------------------------------------------------------------
// Elementwise fp32 -> bf16 (row-major), 4 elems/thread.
// ---------------------------------------------------------------------------
__global__ __launch_bounds__(256) void cvt_bf16(
    const float* __restrict__ x, bf16_t* __restrict__ y, int n4)
{
  const int i = blockIdx.x * 256 + threadIdx.x;
  if (i < n4) {
    const float4 f = reinterpret_cast<const float4*>(x)[i];
    bf16_t* o = y + (size_t)i * 4;
    o[0] = (bf16_t)f.x; o[1] = (bf16_t)f.y;
    o[2] = (bf16_t)f.z; o[3] = (bf16_t)f.w;
  }
}

// ---------------------------------------------------------------------------
// Transpose + convert: W fp32 [K,N] row-major -> Wt bf16 [N,K] row-major.
// One-time cost per weight; makes GEMM B staging a contiguous byte copy.
// ---------------------------------------------------------------------------
__global__ __launch_bounds__(256) void transpose_cvt_bf16(
    const float* __restrict__ W, bf16_t* __restrict__ Wt, int K, int N)
{
  __shared__ float tile[32][33];
  const int tx = threadIdx.x & 31;
  const int ty = threadIdx.x >> 5;         // 0..7
  const int k0 = blockIdx.y * 32;
  const int n0 = blockIdx.x * 32;
  #pragma unroll
  for (int r = ty; r < 32; r += 8) {
    const int k = k0 + r, n = n0 + tx;
    tile[r][tx] = (k < K && n < N) ? W[(size_t)k * N + n] : 0.f;
  }
  __syncthreads();
  #pragma unroll
  for (int r = ty; r < 32; r += 8) {
    const int n = n0 + r, k = k0 + tx;
    if (n < N && k < K) Wt[(size_t)n * K + k] = (bf16_t)tile[tx][r];
  }
}

// ---------------------------------------------------------------------------
// GEMM: C[M,N] = A[M,K] * Bt[N,K]^T; A,Bt bf16 (pre-converted), C fp32.
// Block tile 128x128, 256 threads (8 waves), 8 accumulators of 16x16/wave.
// LDS tiles double-buffered, staged with GLOBAL_LOAD_ASYNC_TO_LDS_B128,
// software-pipelined one K-step ahead. Steady-state loop is branch-free:
// issue(next) -> s_wait_asynccnt 0x4 -> barrier -> 8x WMMA -> barrier.
// ---------------------------------------------------------------------------
#define TM  128
#define TN  128
#define TKK 32

__global__ __launch_bounds__(256) void gemm_bf16(
    const bf16_t* __restrict__ A, const bf16_t* __restrict__ Bt,
    float* __restrict__ C, int M, int N, int Kd)
{
  __shared__ __align__(16) bf16_t As[2][TM][TKK];   // 2 x 8 KB
  __shared__ __align__(16) bf16_t Bs[2][TN][TKK];   // 2 x 8 KB

  const int tid  = threadIdx.x;
  const int lane = tid & 31;
  const int wave = tid >> 5;            // 0..7
  const int m0   = blockIdx.y * TM;
  const int n0   = blockIdx.x * TN;
  const int lan16   = lane & 15;
  const int halfSel = lane >> 4;        // 0: lanes 0-15, 1: lanes 16-31
  const int kb      = halfSel * 8;      // K-base per ISA 16-bit A/B layout
  const unsigned BUFB = (unsigned)(TM * TKK * 2);   // 8 KB buffer stride

  v8f acc[8] = {v8f{}, v8f{}, v8f{}, v8f{}, v8f{}, v8f{}, v8f{}, v8f{}};

  // Loop-invariant staging descriptors: per thread, 2 A chunks + 2 B chunks.
  unsigned lA[2], lB[2];
  const bf16_t* gA[2];
  const bf16_t* gB[2];
  #pragma unroll
  for (int i = 0; i < 2; ++i) {
    const int vlin = tid * 2 + i;       // 0..511
    const int r    = vlin >> 2;         // row 0..127
    const int c8   = (vlin & 3) << 3;   // 8-elem (16 B) chunk
    lA[i] = (unsigned)(size_t)&As[0][r][c8];
    gA[i] = A + (size_t)(m0 + r) * (size_t)Kd + (size_t)c8;
    int ne = n0 + r;                    // clamp ragged N: garbage lands only
    if (ne > N - 1) ne = N - 1;         // in columns discarded at store
    lB[i] = (unsigned)(size_t)&Bs[0][r][c8];
    gB[i] = Bt + (size_t)ne * (size_t)Kd + (size_t)c8;
  }
  auto issue_tile = [&](int k0, int nb) {
    const unsigned off = (unsigned)nb * BUFB;
    #pragma unroll
    for (int i = 0; i < 2; ++i) async_lds_b128(lA[i] + off, gA[i] + k0);
    #pragma unroll
    for (int i = 0; i < 2; ++i) async_lds_b128(lB[i] + off, gB[i] + k0);
  };

  auto compute_tile = [&](int cur) {
    // A fragment: lane m = wave*16+lan16; elems 0-7 <- K kb..kb+7,
    // elems 8-15 <- K kb+16..kb+23 (ISA 16-bit A 16x32 layout)
    const int mrow = wave * 16 + lan16;
    const v8bf alo = *reinterpret_cast<const v8bf*>(&As[cur][mrow][kb]);
    const v8bf ahi = *reinterpret_cast<const v8bf*>(&As[cur][mrow][kb + 16]);
    v16bf afrag;
    #pragma unroll
    for (int i = 0; i < 8; ++i) { afrag[i] = alo[i]; afrag[i + 8] = ahi[i]; }
    #pragma unroll
    for (int j = 0; j < 8; ++j) {
      const int ncol = j * 16 + lan16;  // B: lane -> N, elems -> K
      const v8bf blo = *reinterpret_cast<const v8bf*>(&Bs[cur][ncol][kb]);
      const v8bf bhi = *reinterpret_cast<const v8bf*>(&Bs[cur][ncol][kb + 16]);
      v16bf bfrag;
      #pragma unroll
      for (int i = 0; i < 8; ++i) { bfrag[i] = blo[i]; bfrag[i + 8] = bhi[i]; }
      acc[j] = __builtin_amdgcn_wmma_f32_16x16x32_bf16(
          false, afrag, false, bfrag, (short)0, acc[j], false, false);
    }
  };

  const int NT = Kd / TKK;
  issue_tile(0, 0);

  // steady state: branch-free issue/wait/compute, last tile peeled
  for (int kt = 0; kt < NT - 1; ++kt) {
    const int cur = kt & 1;
    issue_tile((kt + 1) * TKK, cur ^ 1);  // prefetch next tile
    wait_async_le4();                     // oldest 4 (tile kt) have landed
    __syncthreads();                      // async LDS writes visible to all
    compute_tile(cur);
    __syncthreads();                      // done reading before overwrite
  }
  wait_async_le0();                       // last tile
  __syncthreads();
  compute_tile((NT - 1) & 1);

  // store C: elem e -> row m = e + 8*halfSel, lane -> col
  #pragma unroll
  for (int j = 0; j < 8; ++j) {
    const int n = n0 + j * 16 + lan16;
    if (n < N) {
      #pragma unroll
      for (int e = 0; e < 8; ++e) {
        const int m = m0 + wave * 16 + e + halfSel * 8;
        C[(size_t)m * (size_t)N + n] = acc[j][e];
      }
    }
  }
}

// ---------------------------------------------------------------------------
// Elementwise post-QKV: causal depthwise conv (K=4) + SiLU, per-head L2 norm
// of q (scaled by D^-0.5) and k, decay eg = exp(-exp(A_log)*softplus(g+dt)),
// beta = sigmoid(beta_pre). One block per (b,t), 8 channels per thread.
// ---------------------------------------------------------------------------
__global__ __launch_bounds__(256) void postproc(
    const float* __restrict__ qp, const float* __restrict__ kp,
    const float* __restrict__ vp,
    const float* __restrict__ cwq, const float* __restrict__ cwk,
    const float* __restrict__ cwv,
    const float* __restrict__ A_log, const float* __restrict__ dt_bias,
    const float* __restrict__ g_pre, const float* __restrict__ beta_pre,
    float* __restrict__ qo, float* __restrict__ ko, float* __restrict__ vo,
    float* __restrict__ eg, float* __restrict__ beta)
{
  const int bt  = blockIdx.x;
  const int t   = bt % TLEN;
  const int tid = threadIdx.x;
  const int c0  = tid * 8;       // 8 contiguous channels, within one head
  const int h   = tid >> 4;      // 16 threads per head
  __shared__ float red[256];

  float qy[8], ky[8], vy[8];
  #pragma unroll
  for (int i = 0; i < 8; ++i) {
    const int c = c0 + i;
    float aq = 0.f, ak = 0.f, av = 0.f;
    #pragma unroll
    for (int j = 0; j < KC; ++j) {
      const int lag = KC - 1 - j;          // y[t] = sum_j x[t+j-3]*w[j]
      if (t - lag >= 0) {
        const size_t idx = (size_t)(bt - lag) * HIDD + c;
        aq += qp[idx] * cwq[c * KC + j];
        ak += kp[idx] * cwk[c * KC + j];
        av += vp[idx] * cwv[c * KC + j];
      }
    }
    qy[i] = aq / (1.f + expf(-aq));   // SiLU
    ky[i] = ak / (1.f + expf(-ak));
    vy[i] = av / (1.f + expf(-av));
  }

  float ssq = 0.f, ssk = 0.f;
  #pragma unroll
  for (int i = 0; i < 8; ++i) { ssq += qy[i] * qy[i]; ssk += ky[i] * ky[i]; }

  red[tid] = ssq; __syncthreads();
  float totq = 0.f;
  for (int i = 0; i < 16; ++i) totq += red[h * 16 + i];
  __syncthreads();
  red[tid] = ssk; __syncthreads();
  float totk = 0.f;
  for (int i = 0; i < 16; ++i) totk += red[h * 16 + i];

  const float qscale = rsqrtf(totq + EPSV) * 0.08838834764831845f; // *D^-0.5
  const float kscale = rsqrtf(totk + EPSV);
  const float a_h    = expf(A_log[h]);
  const size_t rb    = (size_t)bt * HIDD;

  #pragma unroll
  for (int i = 0; i < 8; ++i) {
    const int c = c0 + i;
    qo[rb + c] = qy[i] * qscale;
    ko[rb + c] = ky[i] * kscale;
    vo[rb + c] = vy[i];
    const float x  = g_pre[rb + c] + dt_bias[c];
    const float sp = (x > 20.f) ? x : log1pf(expf(x));   // softplus, stable
    eg[rb + c] = expf(-a_h * sp);                        // per-step decay
  }
  if (tid < NH) {
    const float bp = beta_pre[(size_t)bt * NH + tid];
    beta[(size_t)bt * NH + tid] = 1.f / (1.f + expf(-bp));
  }
}

// ---------------------------------------------------------------------------
// Gated delta-rule recurrence. One block per (b,h); 256 threads; state
// S[128x128] register-resident. Per-step operands (513 floats) double-
// buffered in LDS via GLOBAL_LOAD_ASYNC_TO_LDS, issued one step ahead.
// ---------------------------------------------------------------------------
__global__ __launch_bounds__(256) void recurrence(
    const float* __restrict__ q, const float* __restrict__ k,
    const float* __restrict__ v, const float* __restrict__ eg,
    const float* __restrict__ beta, float* __restrict__ o)
{
  const int bh  = blockIdx.x;
  const int b   = bh / NH;
  const int h   = bh % NH;
  const int tid = threadIdx.x;
  const int dv  = tid & 127;
  const int half= tid >> 7;
  const int r0  = half * 64;

  // pack: [0..127]=q, [128..255]=k, [256..383]=v, [384..511]=eg, [512]=beta
  __shared__ __align__(16) float buf[2][520];
  __shared__ float part[256];
  __shared__ float delta_s[HD];

  const int e0  = tid * 2;
  const int arr = e0 >> 7;              // 0..3 -> q,k,v,eg
  const int idx = e0 & 127;
  const float* src = (arr == 0) ? q : (arr == 1) ? k : (arr == 2) ? v : eg;
  const float* ga  = src + (size_t)(b * TLEN) * HIDD + (size_t)h * HD + idx;
  const float* gb  = beta + (size_t)(b * TLEN) * NH + h;

  async_lds_b64((unsigned)(size_t)&buf[0][e0], ga);
  if (tid == 0) async_lds_b32((unsigned)(size_t)&buf[0][512], gb);

  float S[64];
  #pragma unroll
  for (int i = 0; i < 64; ++i) S[i] = 0.f;

  for (int t = 0; t < TLEN; ++t) {
    wait_async_le0();       // this step's operand pack landed in LDS
    __syncthreads();        // all waves' async writes visible; buffers free

    if (t + 1 < TLEN) {     // issue next step's pack into the other buffer
      const int par = (t + 1) & 1;
      async_lds_b64((unsigned)(size_t)&buf[par][e0], ga + (size_t)(t + 1) * HIDD);
      if (tid == 0)
        async_lds_b32((unsigned)(size_t)&buf[par][512], gb + (size_t)(t + 1) * NH);
    }

    const float* qs = &buf[t & 1][0];
    const float* ks = qs + 128;
    const float* vs = qs + 256;
    const float* es = qs + 384;
    const float betas = qs[512];

    // decay + k^T S partial
    float ksum = 0.f;
    #pragma unroll
    for (int r = 0; r < 64; ++r) {
      const float s = S[r] * es[r0 + r];
      S[r] = s;
      ksum += ks[r0 + r] * s;
    }
    part[tid] = ksum;
    __syncthreads();
    if (half == 0) {
      const float kS = part[tid] + part[tid + 128];
      delta_s[dv] = betas * (vs[dv] - kS);
    }
    __syncthreads();

    // rank-1 update + q^T S partial
    const float dlt = delta_s[dv];
    float osum = 0.f;
    #pragma unroll
    for (int r = 0; r < 64; ++r) {
      const float s = S[r] + ks[r0 + r] * dlt;
      S[r] = s;
      osum += qs[r0 + r] * s;
    }
    part[tid] = osum;
    __syncthreads();
    if (half == 0) {
      const size_t base = (size_t)(b * TLEN + t) * HIDD + (size_t)h * HD;
      o[base + dv] = part[tid] + part[tid + 128];
    }
    // top-of-loop barrier protects part/delta_s/buf reuse
  }
}

// ---------------------------------------------------------------------------
// Gated RMSNorm over head_dim with sigmoid gate; in-place on o.
// ---------------------------------------------------------------------------
__global__ __launch_bounds__(256) void normgate(
    const float* __restrict__ gate_pre, const float* __restrict__ norm_w,
    float* __restrict__ o)
{
  const int bt  = blockIdx.x;
  const int tid = threadIdx.x;
  const int c0  = tid * 8;
  const int h   = tid >> 4;
  __shared__ float red[256];
  const size_t rb = (size_t)bt * HIDD;

  float ov[8];
  float ss = 0.f;
  #pragma unroll
  for (int i = 0; i < 8; ++i) { ov[i] = o[rb + c0 + i]; ss += ov[i] * ov[i]; }
  red[tid] = ss; __syncthreads();
  float tot = 0.f;
  for (int i = 0; i < 16; ++i) tot += red[h * 16 + i];
  const float scale = rsqrtf(tot / (float)HD + EPSV);

  #pragma unroll
  for (int i = 0; i < 8; ++i) {
    const int c = c0 + i;
    const int d = c & (HD - 1);
    const float gp = gate_pre[rb + c];
    const float sg = 1.f / (1.f + expf(-gp));
    o[rb + c] = ov[i] * scale * norm_w[d] * sg;
  }
}

// ---------------------------------------------------------------------------
extern "C" void kernel_launch(void* const* d_in, const int* in_sizes, int n_in,
                              void* d_out, int out_size, void* d_ws, size_t ws_size,
                              hipStream_t stream)
{
  (void)in_sizes; (void)n_in; (void)out_size; (void)ws_size;

  const float* h       = (const float*)d_in[0];
  const float* Wq      = (const float*)d_in[1];
  const float* Wk      = (const float*)d_in[2];
  const float* Wv      = (const float*)d_in[3];
  const float* cwq     = (const float*)d_in[4];
  const float* cwk     = (const float*)d_in[5];
  const float* cwv     = (const float*)d_in[6];
  const float* A_log   = (const float*)d_in[7];
  const float* dt_bias = (const float*)d_in[8];
  const float* Wfa     = (const float*)d_in[9];
  const float* Wfb     = (const float*)d_in[10];
  const float* Wb      = (const float*)d_in[11];
  const float* Wga     = (const float*)d_in[12];
  const float* Wgb     = (const float*)d_in[13];
  const float* norm_w  = (const float*)d_in[14];
  const float* Wo      = (const float*)d_in[15];

  float* ws = (float*)d_ws;
  const size_t RC = (size_t)BSZ * TLEN * HIDD;        // 8,388,608
  const size_t RD = (size_t)BSZ * TLEN * HD;
  const size_t RH = (size_t)BSZ * TLEN * NH;
  const size_t WW = (size_t)HIDD * HIDD;              // 4,194,304
  const size_t WL = (size_t)HIDD * HD;                // 262,144

  // fp32 region
  float* q_pre    = ws + 0 * RC;
  float* k_pre    = ws + 1 * RC;
  float* v_pre    = ws + 2 * RC;
  float* g_pre    = ws + 3 * RC;
  float* gate_pre = ws + 4 * RC;
  float* qn       = ws + 5 * RC;
  float* kn       = ws + 6 * RC;
  float* vn       = ws + 7 * RC;
  float* eg       = ws + 8 * RC;
  float* o        = ws + 9 * RC;
  float* fa       = ws + 10 * RC;
  float* gaL      = fa + RD;
  float* beta_pre = gaL + RD;
  float* beta     = beta_pre + RH;

  // bf16 region (16B-aligned: fp32 total is a multiple of 4 floats)
  bf16_t* bws   = (bf16_t*)(ws + 10 * RC + 2 * RD + 2 * RH);
  bf16_t* h_bf  = bws;                 bf16_t* o_bf  = h_bf  + RC;
  bf16_t* fa_bf = o_bf + RC;           bf16_t* ga_bf = fa_bf + RD;
  bf16_t* Wq_t  = ga_bf + RD;          bf16_t* Wk_t  = Wq_t + WW;
  bf16_t* Wv_t  = Wk_t + WW;           bf16_t* Wo_t  = Wv_t + WW;
  bf16_t* Wfa_t = Wo_t + WW;           bf16_t* Wfb_t = Wfa_t + WL;
  bf16_t* Wga_t = Wfb_t + WL;          bf16_t* Wgb_t = Wga_t + WL;
  bf16_t* Wb_t  = Wgb_t + WL;

  const int M = BSZ * TLEN;
  dim3 blk(256);
  auto gemm = [&](const bf16_t* A, const bf16_t* Bt, float* C, int m, int n, int kd) {
    dim3 grid((n + TN - 1) / TN, (m + TM - 1) / TM);
    gemm_bf16<<<grid, blk, 0, stream>>>(A, Bt, C, m, n, kd);
  };
  auto cvt = [&](const float* x, bf16_t* y, size_t cnt) {
    const int n4 = (int)(cnt / 4);
    cvt_bf16<<<dim3((n4 + 255) / 256), blk, 0, stream>>>(x, y, n4);
  };
  auto tcvt = [&](const float* W, bf16_t* Wt, int K, int N) {
    dim3 grid((N + 31) / 32, (K + 31) / 32);
    transpose_cvt_bf16<<<grid, blk, 0, stream>>>(W, Wt, K, N);
  };

  // one-time conversions (activations row-major, weights transposed to [N,K])
  cvt(h, h_bf, RC);
  tcvt(Wq,  Wq_t,  HIDD, HIDD);
  tcvt(Wk,  Wk_t,  HIDD, HIDD);
  tcvt(Wv,  Wv_t,  HIDD, HIDD);
  tcvt(Wfa, Wfa_t, HIDD, HD);
  tcvt(Wfb, Wfb_t, HD,   HIDD);
  tcvt(Wb,  Wb_t,  HIDD, NH);
  tcvt(Wga, Wga_t, HIDD, HD);
  tcvt(Wgb, Wgb_t, HD,   HIDD);
  tcvt(Wo,  Wo_t,  HIDD, HIDD);

  // Projections (all bf16 WMMA, async-LDS staged)
  gemm(h_bf, Wq_t,  q_pre,    M, HIDD, HIDD);
  gemm(h_bf, Wk_t,  k_pre,    M, HIDD, HIDD);
  gemm(h_bf, Wv_t,  v_pre,    M, HIDD, HIDD);
  gemm(h_bf, Wfa_t, fa,       M, HD,   HIDD);
  gemm(h_bf, Wga_t, gaL,      M, HD,   HIDD);
  gemm(h_bf, Wb_t,  beta_pre, M, NH,   HIDD);
  cvt(fa,  fa_bf, RD);
  cvt(gaL, ga_bf, RD);
  gemm(fa_bf, Wfb_t, g_pre,    M, HIDD, HD);
  gemm(ga_bf, Wgb_t, gate_pre, M, HIDD, HD);

  // conv + SiLU + l2norm + gates
  postproc<<<dim3(M), blk, 0, stream>>>(q_pre, k_pre, v_pre, cwq, cwk, cwv,
                                        A_log, dt_bias, g_pre, beta_pre,
                                        qn, kn, vn, eg, beta);

  // sequential gated delta-rule scan (async-LDS double-buffered operands)
  recurrence<<<dim3(BSZ * NH), blk, 0, stream>>>(qn, kn, vn, eg, beta, o);

  // gated RMSNorm (in place), then output projection
  normgate<<<dim3(M), blk, 0, stream>>>(gate_pre, norm_w, o);
  cvt(o, o_bf, RC);
  gemm(o_bf, Wo_t, (float*)d_out, M, HIDD, HIDD);
}
